// StableContrastiveLoss_32512902430933
// MI455X (gfx1250) — matrix operand order repaired and checked by reading, not compiled, not run on hardware
//
#include <hip/hip_runtime.h>

// ---------------------------------------------------------------------------
// StableContrastiveLoss for MI455X (gfx1250, wave32, WMMA).
// B=4096, D=512, C=10.  Fused Gram(f16 WMMA) + exp + masked row sums.
// Staging uses CDNA5 async global->LDS copies (ASYNCcnt) with double buffering.
// ---------------------------------------------------------------------------

#define NB    4096
#define DK    512
#define NCLS  10
#define INV_T 14.2857142857142857f   // 1/0.07

#define BM 64              // rows per workgroup
#define BN 64              // cols per inner iteration
#define KC 64              // K chunk staged in LDS
#define NKC (DK / KC)      // 8 K chunks
#define COL_CHUNKS 8       // gridDim.y: column-range split for occupancy
#define COLS_PER_CHUNK (NB / COL_CHUNKS)   // 512
#define LSTR 72            // padded LDS row stride (f16): 144B, 16B aligned, bank-conflict-free

typedef __attribute__((ext_vector_type(16))) _Float16 v16h;
typedef __attribute__((ext_vector_type(8)))  float    v8f;
typedef __attribute__((ext_vector_type(4)))  int      v4i;

typedef __attribute__((address_space(1))) v4i* gv4i_ptr;   // global (AS1) int4*
typedef __attribute__((address_space(3))) v4i* lv4i_ptr;   // LDS    (AS3) int4*

// ---------------- CDNA5 async global->LDS copy helpers ----------------------
// One b128 per lane, LDS written directly by the memory system (no VGPR data),
// tracked with ASYNCcnt. Flat-pointer low 32 bits == LDS byte offset.
__device__ __forceinline__ void async_copy_b128(const void* g, void* l) {
#if __has_builtin(__builtin_amdgcn_global_load_async_to_lds_b128)
  __builtin_amdgcn_global_load_async_to_lds_b128(
      (gv4i_ptr)(unsigned long long)(size_t)g,
      (lv4i_ptr)(unsigned long long)(unsigned)(size_t)l,
      0, 0);
#else
  const unsigned loff = (unsigned)(size_t)l;
  const unsigned long long ga = (unsigned long long)(size_t)g;
  asm volatile("global_load_async_to_lds_b128 %0, %1, off"
               :: "v"(loff), "v"(ga) : "memory");
#endif
}

#if __has_builtin(__builtin_amdgcn_s_wait_asynccnt)
#define WAIT_ASYNC(n) __builtin_amdgcn_s_wait_asynccnt(n)
#else
#define WAIT_ASYNC(n) asm volatile("s_wait_asynccnt %0" :: "i"(n) : "memory")
#endif

// ---------------- Kernel 1: L2-normalize rows -> f16, one-hot -> class id ---
__global__ void prep_kernel(const float* __restrict__ feats,
                            const float* __restrict__ labels,
                            _Float16* __restrict__ f16out,
                            int* __restrict__ cls) {
  const int wave = threadIdx.x >> 5;
  const int lane = threadIdx.x & 31;
  const int row  = blockIdx.x * 8 + wave;       // one wave per row
  const float* src = feats + (size_t)row * DK;
  float v[16];
  float ss = 0.f;
#pragma unroll
  for (int i = 0; i < 16; ++i) {
    v[i] = src[lane + i * 32];
    ss += v[i] * v[i];
  }
#pragma unroll
  for (int off = 16; off >= 1; off >>= 1) ss += __shfl_xor(ss, off, 32);
  const float rn = rsqrtf(ss);
  _Float16* dst = f16out + (size_t)row * DK;
#pragma unroll
  for (int i = 0; i < 16; ++i) dst[lane + i * 32] = (_Float16)(v[i] * rn);

  if (lane == 0) {
    int c = 0;
    const float* lab = labels + row * NCLS;
    for (int j = 0; j < NCLS; ++j)
      if (lab[j] > 0.5f) c = j;      // one-hot -> class id
    cls[row] = c;
  }
}

// ---------------- Kernel 2: fused WMMA Gram + exp + masked row sums ---------
// Block: 64 rows x 512-col chunk. 8 waves; wave = (row tile rt 0..3,
// column half ch 0..1 -> col tiles 2ch, 2ch+1). LDS double-buffered; each
// thread issues 4 async b128 copies per K chunk, next chunk overlaps WMMA.
__global__ __launch_bounds__(256) void gram_kernel(
    const _Float16* __restrict__ feats,   // [NB][DK], L2-normalized f16
    const int* __restrict__ cls,
    float* __restrict__ posArr, float* __restrict__ allArr) {
  __shared__ _Float16 lds[2][(BM + BN) * LSTR];   // A tile then B tile, ~36 KB

  const int tid  = threadIdx.x;
  const int lane = tid & 31;
  const int wave = tid >> 5;
  const int rt   = wave & 3;            // row tile within block
  const int ch   = wave >> 2;           // column half
  const int rowBase = blockIdx.x * BM;
  const int rhalf   = (lane >> 4) << 3; // C-layout: lanes 16-31 hold rows +8

  // Per-thread staging coordinates: 512 b128 segments per tile, 2 per thread.
  const int r0 = tid >> 3,          s0 = tid & 7;
  const int r1 = (tid + 256) >> 3,  s1 = tid & 7;   // (tid+256)&7 == tid&7

  // Per-lane fixed rows (C/D layout: VGPR r -> row rt*16 + rhalf + r)
  int rowg[8], rcls[8];
#pragma unroll
  for (int r = 0; r < 8; ++r) {
    rowg[r] = rowBase + rt * 16 + rhalf + r;
    rcls[r] = cls[rowg[r]];
  }
  float accPos[8], accAll[8];
#pragma unroll
  for (int r = 0; r < 8; ++r) { accPos[r] = 0.f; accAll[r] = 0.f; }

  for (int cb = 0; cb < COLS_PER_CHUNK / BN; ++cb) {
    const int colBase = blockIdx.y * COLS_PER_CHUNK + cb * BN;
    v8f c0 = {}; v8f c1 = {};

    // Issue one K-chunk stage (4 async b128 per thread: 2 A segs + 2 B segs).
    auto issue_stage = [&](int k0, _Float16* buf) {
      _Float16* bufB = buf + BM * LSTR;
      async_copy_b128(feats + (size_t)(rowBase + r0) * DK + k0 + s0 * 8,
                      buf  + r0 * LSTR + s0 * 8);
      async_copy_b128(feats + (size_t)(colBase + r0) * DK + k0 + s0 * 8,
                      bufB + r0 * LSTR + s0 * 8);
      async_copy_b128(feats + (size_t)(rowBase + r1) * DK + k0 + s1 * 8,
                      buf  + r1 * LSTR + s1 * 8);
      async_copy_b128(feats + (size_t)(colBase + r1) * DK + k0 + s1 * 8,
                      bufB + r1 * LSTR + s1 * 8);
    };

    issue_stage(0, &lds[0][0]);                       // prologue: kc = 0

    for (int kc = 0; kc < NKC; ++kc) {
      _Float16* curA = &lds[kc & 1][0];
      _Float16* curB = curA + BM * LSTR;

      if (kc + 1 < NKC) {
        // Kick off next chunk into the other buffer (its last readers
        // finished before the trailing barrier of iteration kc-1).
        issue_stage((kc + 1) * KC, &lds[(kc + 1) & 1][0]);
        WAIT_ASYNC(4);   // retire the 4 older copies (async loads done in order)
      } else {
        WAIT_ASYNC(0);
      }
      __syncthreads();   // chunk kc visible to all waves

#pragma unroll
      for (int ks = 0; ks < KC; ks += 32) {
        // A frag (ISA 7.12.2, 16-bit A 16x32): lanes 0-15 -> K {ks..+7, ks+16..+23};
        // lanes 16-31 same pattern shifted +8.
        const _Float16* pa = curA + (rt * 16 + (lane & 15)) * LSTR;
        const int ka = ks + ((lane >> 4) << 3);
        v16h a;
        ((uint4*)&a)[0] = *(const uint4*)(pa + ka);
        ((uint4*)&a)[1] = *(const uint4*)(pa + ka + 16);
        // B frag: lane n holds column n, K {kb..+7, kb+16..+23}, kb = ks + 8*(lane>=16)
        const int kb = ks + ((lane >> 4) << 3);
        const _Float16* pb0 = curB + ((ch * 2 + 0) * 16 + (lane & 15)) * LSTR;
        const _Float16* pb1 = curB + ((ch * 2 + 1) * 16 + (lane & 15)) * LSTR;
        v16h b0, b1;
        ((uint4*)&b0)[0] = *(const uint4*)(pb0 + kb);
        ((uint4*)&b0)[1] = *(const uint4*)(pb0 + kb + 16);
        ((uint4*)&b1)[0] = *(const uint4*)(pb1 + kb);
        ((uint4*)&b1)[1] = *(const uint4*)(pb1 + kb + 16);

        c0 = __builtin_amdgcn_wmma_f32_16x16x32_f16(false, a, false, b0,
                                                    (short)0, c0, false, false);
        c1 = __builtin_amdgcn_wmma_f32_16x16x32_f16(false, a, false, b1,
                                                    (short)0, c1, false, false);
      }
      __syncthreads();   // all reads of buffer (kc&1) done before its reuse
    }

    // Fused epilogue: clip/T, exp, diagonal mask, class-masked accumulation.
    const int col0 = colBase + (ch * 2 + 0) * 16 + (lane & 15);
    const int col1 = colBase + (ch * 2 + 1) * 16 + (lane & 15);
    const int ccls0 = cls[col0];
    const int ccls1 = cls[col1];
#pragma unroll
    for (int r = 0; r < 8; ++r) {
      float s0v = fminf(fmaxf(c0[r] * INV_T, -20.f), 20.f);
      float s1v = fminf(fmaxf(c1[r] * INV_T, -20.f), 20.f);
      float e0 = (rowg[r] == col0) ? 0.f : __expf(s0v);   // diag -> 0
      float e1 = (rowg[r] == col1) ? 0.f : __expf(s1v);
      accAll[r] += e0 + e1;
      if (rcls[r] == ccls0) accPos[r] += e0;
      if (rcls[r] == ccls1) accPos[r] += e1;
    }
  }

  // Reduce across the 16 column-lanes of each half; lanes 0 and 16 hold
  // disjoint row sets -> two atomic writers per wave per r.
#pragma unroll
  for (int r = 0; r < 8; ++r) {
    float p = accPos[r], a = accAll[r];
#pragma unroll
    for (int off = 8; off >= 1; off >>= 1) {
      p += __shfl_xor(p, off, 16);
      a += __shfl_xor(a, off, 16);
    }
    if ((lane & 15) == 0) {
      atomicAdd(&posArr[rowg[r]], p);
      atomicAdd(&allArr[rowg[r]], a);
    }
  }
}

// ---------------- Kernel 3: per-row loss + valid count ----------------------
__global__ void rowloss_kernel(const float* __restrict__ posArr,
                               const float* __restrict__ allArr,
                               float* __restrict__ accum) {
  const int row = blockIdx.x * blockDim.x + threadIdx.x;
  const float p = posArr[row];
  const float a = allArr[row];
  const bool valid = p > 0.f;                       // any positive pair
  float l = valid ? -logf(p / (a + 1e-8f) + 1e-8f) : 0.f;
  float c = valid ? 1.f : 0.f;
#pragma unroll
  for (int off = 16; off >= 1; off >>= 1) {
    l += __shfl_xor(l, off, 32);
    c += __shfl_xor(c, off, 32);
  }
  if ((threadIdx.x & 31) == 0) {
    atomicAdd(&accum[0], l);
    atomicAdd(&accum[1], c);
  }
}

// ---------------- Kernel 4: final scalar ------------------------------------
__global__ void finalize_kernel(const float* __restrict__ accum,
                                float* __restrict__ out) {
  out[0] = (accum[1] > 0.f) ? accum[0] / accum[1] : 0.f;
}

// ---------------------------------------------------------------------------
extern "C" void kernel_launch(void* const* d_in, const int* in_sizes, int n_in,
                              void* d_out, int out_size, void* d_ws, size_t ws_size,
                              hipStream_t stream) {
  const float* feats  = (const float*)d_in[0];   // [4096,512] f32
  const float* labels = (const float*)d_in[1];   // [4096,10]  f32 one-hot

  char* ws = (char*)d_ws;
  _Float16* f16   = (_Float16*)ws;                           // 4 MB
  int*      cls   = (int*)(ws + (size_t)NB * DK * 2);        // 16 KB
  float*    posA  = (float*)(ws + (size_t)NB * DK * 2 + NB * 4);
  float*    allA  = posA + NB;
  float*    accum = allA + NB;                               // 2 floats

  // Zero posArr/allArr/accum in one contiguous async memset (capture-safe).
  (void)hipMemsetAsync(posA, 0, (size_t)(2 * NB + 2) * sizeof(float), stream);

  prep_kernel<<<NB / 8, 256, 0, stream>>>(feats, labels, f16, cls);
  gram_kernel<<<dim3(NB / BM, COL_CHUNKS), 256, 0, stream>>>(f16, cls, posA, allA);
  rowloss_kernel<<<NB / 256, 256, 0, stream>>>(posA, allA, accum);
  finalize_kernel<<<1, 1, 0, stream>>>(accum, (float*)d_out);
}
